// AnyNetRefinement_73933567033728
// MI455X (gfx1250) — compile-verified, matrix operand order checked by reading, not run on hardware
//
#include <hip/hip_runtime.h>
#include <hip/hip_bf16.h>

// ---------------------------------------------------------------------------
// AnyNet disparity refinement for MI455X (gfx1250, wave32).
//   conv2/conv3 (16->16) and conv4 (16->24) are implicit-GEMM bf16 WMMA
//   kernels: one wave computes a 16-pixel x 16-outchannel tile,
//   K = 9 taps * 16 cin padded to 10 taps -> 5 v_wmma_f32_16x16x32_bf16.
//   Activations live in NHWC bf16 so each A-fragment half is one b128 load.
//   BN+ReLU fused in registers; D tile staged through LDS for coalesced
//   b128 stores; conv4 fuses the guidance normalization + transpose.
// ---------------------------------------------------------------------------

typedef __attribute__((ext_vector_type(16))) __bf16 v16bf;
typedef __attribute__((ext_vector_type(8)))  __bf16 v8bf;
typedef __attribute__((ext_vector_type(8)))  float  v8f;

#define B_  8
#define H_  384
#define W_  640
#define P_  8
#define TILES_X (W_ / 16)                 // 40
#define NWAVE   (B_ * H_ * TILES_X)       // 122880 tiles total
#define NBLK    (NWAVE / 8)               // 8 waves / block

__device__ __forceinline__ v8bf zero_v8bf() {
    v8bf z;
#pragma unroll
    for (int i = 0; i < 8; ++i) z[i] = (__bf16)0.0f;
    return z;
}

// ---------------------------------------------------------------------------
// conv1: 3 -> 16 direct conv + BN + ReLU, writes NHWC bf16. (K=27, tiny)
// ---------------------------------------------------------------------------
__global__ __launch_bounds__(256)
void conv1_kernel(const float* __restrict__ img, const float* __restrict__ w1,
                  const float* __restrict__ g,  const float* __restrict__ bt,
                  const float* __restrict__ mn, const float* __restrict__ vr,
                  __bf16* __restrict__ out) {
    int idx = blockIdx.x * blockDim.x + threadIdx.x;
    if (idx >= B_ * H_ * W_) return;
    int x = idx % W_;
    int y = (idx / W_) % H_;
    int b = idx / (W_ * H_);

    float acc[16];
#pragma unroll
    for (int o = 0; o < 16; ++o) acc[o] = 0.0f;

#pragma unroll
    for (int ky = 0; ky < 3; ++ky) {
        int iy = y + ky - 1;
        bool vy = (iy >= 0) && (iy < H_);
#pragma unroll
        for (int kx = 0; kx < 3; ++kx) {
            int ix = x + kx - 1;
            bool ok = vy && (ix >= 0) && (ix < W_);
#pragma unroll
            for (int ci = 0; ci < 3; ++ci) {
                float pv = ok ? img[((b * 3 + ci) * H_ + iy) * W_ + ix] : 0.0f;
#pragma unroll
                for (int o = 0; o < 16; ++o)
                    acc[o] = fmaf(pv, w1[((o * 3 + ci) * 3 + ky) * 3 + kx], acc[o]);
            }
        }
    }
    v8bf lo = zero_v8bf(), hi = zero_v8bf();
#pragma unroll
    for (int o = 0; o < 16; ++o) {
        float s  = g[o] * rsqrtf(vr[o] + 1e-5f);
        float bo = bt[o] - mn[o] * s;
        float v  = fmaxf(acc[o] * s + bo, 0.0f);
        if (o < 8) lo[o] = (__bf16)v; else hi[o - 8] = (__bf16)v;
    }
    *(v8bf*)&out[(size_t)idx * 16]     = lo;
    *(v8bf*)&out[(size_t)idx * 16 + 8] = hi;
}

// ---------------------------------------------------------------------------
// conv2/conv3: 16 -> 16 implicit-GEMM WMMA + BN + ReLU, NHWC bf16 in/out.
// Wave tile: pixels (b, y, x0..x0+15) x outchannels 0..15.
// ---------------------------------------------------------------------------
__global__ __launch_bounds__(256)
void conv16_wmma(const __bf16* __restrict__ in, const float* __restrict__ wt,
                 const float* __restrict__ g,  const float* __restrict__ bt,
                 const float* __restrict__ mn, const float* __restrict__ vr,
                 __bf16* __restrict__ out) {
    __shared__ alignas(16) __bf16 stg[8][16][16];

    int wid  = threadIdx.x >> 5;
    int lane = threadIdx.x & 31;
    int gw   = blockIdx.x * 8 + wid;               // tile id, exact cover
    int xt = gw % TILES_X;
    int y  = (gw / TILES_X) % H_;
    int b  = gw / (TILES_X * H_);
    int x0 = xt * 16;

    int oc    = lane & 15;                         // D column / out channel
    int half  = lane >> 4;                         // K-half select (wave32 layout)
    int choff = half * 8;                          // input-channel sub-block
    int mrow  = lane & 15;                         // A row = pixel within tile

    // Per-lane fused-BN constants for channel `oc`.
    float sc = g[oc] * rsqrtf(vr[oc] + 1e-5f);
    float bs = bt[oc] - mn[oc] * sc;

    // Build 5 weight (B) fragments: tap pairs (0,1)(2,3)(4,5)(6,7)(8,pad).
    // Lane<16 supplies K 0..7 & 16..23; lane>=16 supplies K 8..15 & 24..31.
    v16bf bw[5];
#pragma unroll
    for (int pair = 0; pair < 5; ++pair) {
#pragma unroll
        for (int e = 0; e < 16; ++e) {
            int tap = pair * 2 + (e >> 3);
            float wv = 0.0f;
            if (tap < 9) {
                int ky = tap / 3, kx = tap % 3;
                int ci = choff + (e & 7);
                wv = wt[((oc * 16 + ci) * 3 + ky) * 3 + kx];
            }
            bw[pair][e] = (__bf16)wv;
        }
    }

    v8f acc;
#pragma unroll
    for (int r = 0; r < 8; ++r) acc[r] = 0.0f;

#pragma unroll
    for (int pair = 0; pair < 5; ++pair) {
        union { v16bf v; v8bf h[2]; } af;
#pragma unroll
        for (int t = 0; t < 2; ++t) {
            int tap = pair * 2 + t;
            v8bf val = zero_v8bf();
            if (tap < 9) {
                int ky = tap / 3, kx = tap % 3;
                int iy = y + ky - 1;
                int ix = x0 + mrow + kx - 1;
                if (iy >= 0 && iy < H_ && ix >= 0 && ix < W_)
                    val = *(const v8bf*)&in[(size_t)((b * H_ + iy) * W_ + ix) * 16 + choff];
            }
            af.h[t] = val;
        }
        acc = __builtin_amdgcn_wmma_f32_16x16x32_bf16(
                  false, af.v, false, bw[pair], (short)0, acc, false, false);
    }

    // BN + ReLU, stage to LDS (D layout: VGPR r -> M = r + 8*half, N = oc).
#pragma unroll
    for (int r = 0; r < 8; ++r) {
        float v = fmaxf(acc[r] * sc + bs, 0.0f);
        stg[wid][r + half * 8][oc] = (__bf16)v;
    }
    __syncthreads();

    // Coalesced b128 store: lane -> pixel row mrow, channels choff..choff+7.
    v8bf ov = *(const v8bf*)&stg[wid][mrow][choff];
    *(v8bf*)&out[(size_t)((b * H_ + y) * W_ + x0 + mrow) * 16 + choff] = ov;
}

// ---------------------------------------------------------------------------
// conv4: 16 -> 24 WMMA (two N-tiles, second zero-padded to 32), fused
// |G1|+|G2|+|G3| normalization + transpose to SPN layout [b*P+p][x][y].
// ---------------------------------------------------------------------------
__global__ __launch_bounds__(256)
void conv4_wmma(const __bf16* __restrict__ in, const float* __restrict__ w4,
                float* __restrict__ cA, float* __restrict__ cB, float* __restrict__ cC) {
    __shared__ float gst[8][16][24];

    int wid  = threadIdx.x >> 5;
    int lane = threadIdx.x & 31;
    int gw   = blockIdx.x * 8 + wid;
    int xt = gw % TILES_X;
    int y  = (gw / TILES_X) % H_;
    int b  = gw / (TILES_X * H_);
    int x0 = xt * 16;

    int oc    = lane & 15;
    int half  = lane >> 4;
    int choff = half * 8;
    int mrow  = lane & 15;

    v16bf bw0[5], bw1[5];
#pragma unroll
    for (int pair = 0; pair < 5; ++pair) {
#pragma unroll
        for (int e = 0; e < 16; ++e) {
            int tap = pair * 2 + (e >> 3);
            float w0 = 0.0f, w1v = 0.0f;
            if (tap < 9) {
                int ky = tap / 3, kx = tap % 3;
                int ci = choff + (e & 7);
                w0 = w4[((oc * 16 + ci) * 3 + ky) * 3 + kx];        // ch 0..15
                int o2 = 16 + oc;                                    // ch 16..23 (+pad)
                if (o2 < 24)
                    w1v = w4[((o2 * 16 + ci) * 3 + ky) * 3 + kx];
            }
            bw0[pair][e] = (__bf16)w0;
            bw1[pair][e] = (__bf16)w1v;
        }
    }

    v8f acc0, acc1;
#pragma unroll
    for (int r = 0; r < 8; ++r) { acc0[r] = 0.0f; acc1[r] = 0.0f; }

#pragma unroll
    for (int pair = 0; pair < 5; ++pair) {
        union { v16bf v; v8bf h[2]; } af;
#pragma unroll
        for (int t = 0; t < 2; ++t) {
            int tap = pair * 2 + t;
            v8bf val = zero_v8bf();
            if (tap < 9) {
                int ky = tap / 3, kx = tap % 3;
                int iy = y + ky - 1;
                int ix = x0 + mrow + kx - 1;
                if (iy >= 0 && iy < H_ && ix >= 0 && ix < W_)
                    val = *(const v8bf*)&in[(size_t)((b * H_ + iy) * W_ + ix) * 16 + choff];
            }
            af.h[t] = val;
        }
        acc0 = __builtin_amdgcn_wmma_f32_16x16x32_bf16(
                   false, af.v, false, bw0[pair], (short)0, acc0, false, false);
        acc1 = __builtin_amdgcn_wmma_f32_16x16x32_bf16(
                   false, af.v, false, bw1[pair], (short)0, acc1, false, false);
    }

    // Stage raw G (24 channels x 16 pixels) in LDS.
#pragma unroll
    for (int r = 0; r < 8; ++r) {
        int M = r + half * 8;
        gst[wid][M][oc] = acc0[r];
        if (oc < 8) gst[wid][M][16 + oc] = acc1[r];
    }
    __syncthreads();

    // Normalize + transpose: 16 pixels * 8 p = 128 triples, 4 per lane.
#pragma unroll
    for (int j = 0; j < 4; ++j) {
        int t   = lane * 4 + j;
        int pix = t >> 3;
        int p   = t & 7;
        float g1 = gst[wid][pix][p];
        float g2 = gst[wid][pix][p + 8];
        float g3 = gst[wid][pix][p + 16];
        float sa = fabsf(g1) + fabsf(g2) + fabsf(g3) + 1e-8f;
        size_t o = ((size_t)(b * P_ + p) * W_ + (x0 + pix)) * H_ + y;
        cA[o] = g1 / sa;
        cB[o] = g2 / sa;
        cC[o] = g3 / sa;
    }
}

// ---------------------------------------------------------------------------
// convd: disp (1ch) -> 8 features, written directly in SPN layout [b*P+p][x][y].
// ---------------------------------------------------------------------------
__global__ __launch_bounds__(256)
void convd_kernel(const float* __restrict__ disp, const float* __restrict__ wd,
                  float* __restrict__ dfeat) {
    int idx = blockIdx.x * blockDim.x + threadIdx.x;
    if (idx >= B_ * H_ * W_) return;
    int x = idx % W_;
    int y = (idx / W_) % H_;
    int b = idx / (W_ * H_);

    float pv[9];
#pragma unroll
    for (int ky = 0; ky < 3; ++ky) {
        int iy = y + ky - 1;
#pragma unroll
        for (int kx = 0; kx < 3; ++kx) {
            int ix = x + kx - 1;
            bool ok = (iy >= 0) && (iy < H_) && (ix >= 0) && (ix < W_);
            pv[ky * 3 + kx] = ok ? disp[(b * H_ + iy) * W_ + ix] : 0.0f;
        }
    }
#pragma unroll
    for (int p = 0; p < P_; ++p) {
        float s = 0.0f;
#pragma unroll
        for (int t = 0; t < 9; ++t) s = fmaf(pv[t], wd[p * 9 + t], s);
        dfeat[((size_t)(b * P_ + p) * W_ + x) * H_ + y] = s;
    }
}

// ---------------------------------------------------------------------------
// SPN left->right scan over W. One block per (b,p) plane, 384 threads = H.
// hn[y] = (1-a-b-c)*x[y] + a*h[y-1] + b*h[y] + c*h[y+1] ; LDS halo exchange.
// ---------------------------------------------------------------------------
__global__ __launch_bounds__(384)
void spn_kernel(const float* __restrict__ cA, const float* __restrict__ cB,
                const float* __restrict__ cC, const float* __restrict__ xf,
                float* __restrict__ prop) {
    __shared__ float hbuf[H_ + 2];
    int tid = threadIdx.x;                     // y
    size_t base = (size_t)blockIdx.x * W_ * H_;
    if (tid == 0) { hbuf[0] = 0.0f; hbuf[H_ + 1] = 0.0f; }
    float h = 0.0f;
    for (int t = 0; t < W_; ++t) {
        size_t o = base + (size_t)t * H_ + tid;
        float a  = cA[o], bb = cB[o], c = cC[o], xv = xf[o];
        hbuf[tid + 1] = h;
        __syncthreads();
        float up = hbuf[tid];
        float dn = hbuf[tid + 2];
        float hn = (1.0f - a - bb - c) * xv + a * up + bb * h + c * dn;
        prop[o] = hn;
        h = hn;
        __syncthreads();                       // protect hbuf for next step
    }
}

// ---------------------------------------------------------------------------
// refine: res = conv(prop, wc) (8->1), out = relu(res + disp).
// ---------------------------------------------------------------------------
__global__ __launch_bounds__(256)
void refine_kernel(const float* __restrict__ prop, const float* __restrict__ wc,
                   const float* __restrict__ disp, float* __restrict__ out) {
    int idx = blockIdx.x * blockDim.x + threadIdx.x;
    if (idx >= B_ * H_ * W_) return;
    int x = idx % W_;
    int y = (idx / W_) % H_;
    int b = idx / (W_ * H_);

    float acc = 0.0f;
#pragma unroll
    for (int p = 0; p < P_; ++p) {
        size_t pb = (size_t)(b * P_ + p) * W_ * H_;
#pragma unroll
        for (int ky = 0; ky < 3; ++ky) {
            int iy = y + ky - 1;
#pragma unroll
            for (int kx = 0; kx < 3; ++kx) {
                int ix = x + kx - 1;
                if (iy >= 0 && iy < H_ && ix >= 0 && ix < W_)
                    acc = fmaf(prop[pb + (size_t)ix * H_ + iy],
                               wc[(p * 3 + ky) * 3 + kx], acc);
            }
        }
    }
    out[idx] = fmaxf(acc + disp[idx], 0.0f);
}

// ---------------------------------------------------------------------------
extern "C" void kernel_launch(void* const* d_in, const int* in_sizes, int n_in,
                              void* d_out, int out_size, void* d_ws, size_t ws_size,
                              hipStream_t stream) {
    const float* disp   = (const float*)d_in[0];
    // d_in[1]=left, d_in[2]=right, d_in[4]=rightImage: unused by reference()
    const float* limg   = (const float*)d_in[3];
    const float* w1     = (const float*)d_in[5];
    const float* w2     = (const float*)d_in[6];
    const float* w3     = (const float*)d_in[7];
    const float* w4     = (const float*)d_in[8];
    const float* wd     = (const float*)d_in[9];
    const float* wc     = (const float*)d_in[10];
    const float* bn1g = (const float*)d_in[11], *bn1b = (const float*)d_in[12];
    const float* bn1m = (const float*)d_in[13], *bn1v = (const float*)d_in[14];
    const float* bn2g = (const float*)d_in[15], *bn2b = (const float*)d_in[16];
    const float* bn2m = (const float*)d_in[17], *bn2v = (const float*)d_in[18];
    const float* bn3g = (const float*)d_in[19], *bn3b = (const float*)d_in[20];
    const float* bn3m = (const float*)d_in[21], *bn3v = (const float*)d_in[22];

    // Workspace: 7 equal 62.9 MB chunks (bf16 act = f32 plane byte size).
    const size_t CHUNK = (size_t)B_ * H_ * W_ * 16 * 2;   // 62,914,560 B
    char* ws = (char*)d_ws;
    __bf16* bufA  = (__bf16*)(ws + 0 * CHUNK);   // x1, then x3
    __bf16* bufB  = (__bf16*)(ws + 1 * CHUNK);   // x2
    float*  coefA = (float*)(ws + 2 * CHUNK);
    float*  coefB = (float*)(ws + 3 * CHUNK);
    float*  coefC = (float*)(ws + 4 * CHUNK);
    float*  dfeat = (float*)(ws + 5 * CHUNK);
    float*  prop  = (float*)(ws + 6 * CHUNK);

    const int npix = B_ * H_ * W_;
    const int eb   = (npix + 255) / 256;

    conv1_kernel<<<eb, 256, 0, stream>>>(limg, w1, bn1g, bn1b, bn1m, bn1v, bufA);
    conv16_wmma<<<NBLK, 256, 0, stream>>>(bufA, w2, bn2g, bn2b, bn2m, bn2v, bufB);
    conv16_wmma<<<NBLK, 256, 0, stream>>>(bufB, w3, bn3g, bn3b, bn3m, bn3v, bufA);
    conv4_wmma<<<NBLK, 256, 0, stream>>>(bufA, w4, coefA, coefB, coefC);
    convd_kernel<<<eb, 256, 0, stream>>>(disp, wd, dfeat);
    spn_kernel<<<B_ * P_, 384, 0, stream>>>(coefA, coefB, coefC, dfeat, prop);
    refine_kernel<<<eb, 256, 0, stream>>>(prop, wc, disp, (float*)d_out);
}
